// VSMLRNN_32117765439683
// MI455X (gfx1250) — compile-verified, matrix operand order checked by reading, not run on hardware
//
#include <hip/hip_runtime.h>
#include <hip/hip_bf16.h>

typedef __attribute__((ext_vector_type(16))) _Float16 v16h;
typedef __attribute__((ext_vector_type(8)))  _Float16 v8h;
typedef __attribute__((ext_vector_type(8)))  float    v8f;

#define SLOW 32
#define MSG  16
#define MICRO 2
#define TT   4
#define BB   8
#define NI0  256
#define NO0  128
#define NI1  128
#define NO1  64
#define NOUT 64
#define EPSV 1e-6f

// Branch-free activations (no EXEC-divergent libm paths inside the WMMA loop).
__device__ __forceinline__ float sigf(float x){
  return __builtin_amdgcn_rcpf(1.0f + __expf(-x));
}
__device__ __forceinline__ float tanh_fast(float x){
  // tanh(x) = 1 - 2/(exp(2x)+1); exp saturates cleanly for |x| large.
  return 1.0f - 2.0f*__builtin_amdgcn_rcpf(__expf(2.0f*x) + 1.0f);
}

__device__ __forceinline__ v8f wmma16(v16h a, v16h b, v8f c){
  return __builtin_amdgcn_wmma_f32_16x16x32_f16(false, a, false, b, (short)0, c, false, false);
}

// Load a WMMA A/B fragment for this lane from an f16 [row][32] block.
// lanes 0-15: row=lane,    K = 0..7  and 16..23
// lanes16-31: row=lane-16, K = 8..15 and 24..31
__device__ __forceinline__ v16h frag_from_rows(const _Float16* base, int r, int half){
  const _Float16* p = base + r*32;
  v8h lo = *(const v8h*)(p + half*8);
  v8h hi = *(const v8h*)(p + 16 + half*8);
  v16h out;
  #pragma unroll
  for (int i=0;i<8;i++){ out[i]=lo[i]; out[8+i]=hi[i]; }
  return out;
}

__global__ void zero_kernel(float* p, size_t n){
  size_t i = (size_t)blockIdx.x*blockDim.x + threadIdx.x;
  size_t stride = (size_t)gridDim.x*blockDim.x;
  for (; i<n; i+=stride) p[i] = 0.0f;
}

// _merge: first half of SLOW state averaged over batch, broadcast back.
__global__ void merge_kernel(float* s, int cells){
  int idx = blockIdx.x*blockDim.x + threadIdx.x;
  int total = cells * (SLOW/2);
  if (idx >= total) return;
  int k    = idx & 15;
  int cell = idx >> 4;
  size_t stride = (size_t)cells * SLOW;
  size_t off    = (size_t)cell*SLOW + k;
  float sum = 0.0f;
  #pragma unroll
  for (int b=0;b<BB;b++) sum += s[(size_t)b*stride + off];
  float m = sum * (1.0f/BB);
  #pragma unroll
  for (int b=0;b<BB;b++) s[(size_t)b*stride + off] = m;
}

// fwd0 <- pad(input), fwd1 <- 0, outacc <- 0
__global__ void prep_kernel(const float* __restrict__ inp_t, float* fwd0,
                            float* fwd1, float* outacc){
  const int N0 = BB*NI0*MSG;   // 32768
  const int N1 = BB*NO0*MSG;   // 16384
  const int N2 = BB*NOUT;      // 512
  int idx = blockIdx.x*blockDim.x + threadIdx.x;
  if (idx < N0){
    int k = idx & 15;
    fwd0[idx] = (k==0) ? inp_t[idx>>4] : 0.0f;
  } else if (idx < N0+N1){
    fwd1[idx-N0] = 0.0f;
  } else if (idx < N0+N1+N2){
    outacc[idx-N0-N1] = 0.0f;
  }
}

// One wave processes 16-cell tiles (16 consecutive o for fixed b,i).
template<bool LAST>
__global__ void __launch_bounds__(256, 1) layer_kernel(
    const float* __restrict__ Wi,  const float* __restrict__ Wh,
    const float* __restrict__ bl,
    const float* __restrict__ Wf,  const float* __restrict__ bf,
    const float* __restrict__ Wb,  const float* __restrict__ bwbp,
    const float* __restrict__ lnfs, const float* __restrict__ lnfb,
    const float* __restrict__ lnbs, const float* __restrict__ lnbb,
    const float* __restrict__ fwd, const float* __restrict__ bwd,
    float* __restrict__ hst, float* __restrict__ cst,
    float* __restrict__ accF, float* __restrict__ accB,
    int NI, int NO, float ascF, float ascB, int tiles_per_wave)
{
  __shared__ __align__(16) _Float16 sWi[128*32];  // transposed [n][k]
  __shared__ __align__(16) _Float16 sWh[128*32];
  __shared__ __align__(16) _Float16 sWf[16*32];
  __shared__ __align__(16) _Float16 sWb[16*32];
  __shared__ __align__(16) _Float16 sH[8][16*32]; // per-wave h staging (A layout src)
  __shared__ __align__(16) float    sL[8][512];   // per-wave LN staging fm|bm

  const int tid = threadIdx.x;
  for (int idx = tid; idx < 128*32; idx += 256){
    int n = idx & 127, k = idx >> 7;
    sWi[n*32+k] = (_Float16)Wi[idx];
    sWh[n*32+k] = (_Float16)Wh[idx];
  }
  for (int idx = tid; idx < 16*32; idx += 256){
    int n = idx & 15, k = idx >> 4;
    sWf[n*32+k] = (_Float16)Wf[idx];
    sWb[n*32+k] = (_Float16)Wb[idx];
  }
  __syncthreads();

  const int lane = tid & 31;
  const int wid  = tid >> 5;
  const int half = lane >> 4;
  const int r    = lane & 15;
  const int nob  = NO >> 4;
  const int total = BB * NI * nob;
  const int gw = blockIdx.x * 8 + wid;

  float biasg[8];
  #pragma unroll
  for (int t=0;t<8;t++) biasg[t] = bl[16*t + r];
  const float biasf = bf[r];
  const float biasb = LAST ? bwbp[r] : 0.0f;

  int tend = (gw+1)*tiles_per_wave; if (tend > total) tend = total;
  for (int tile = gw*tiles_per_wave; tile < tend; ++tile){
    const int ob  = tile % nob;
    const int tmp = tile / nob;
    const int i   = tmp % NI;
    const int b   = tmp / NI;
    const int o0  = ob*16;
    const size_t cellbase = ((size_t)(b*NI + i))*NO + o0;

    // ---- A_x: K 0..15 = fwd msg (same for all rows), K 16..31 = bwd msg (per o)
    v16h ax;
    {
      const float* fp = fwd + ((size_t)(b*NI + i))*MSG + half*8;
      const float* bp = bwd + ((size_t)(b*NO) + o0 + r)*MSG + half*8;
      #pragma unroll
      for (int k=0;k<8;k++){ ax[k] = (_Float16)fp[k]; ax[8+k] = (_Float16)bp[k]; }
    }

    // ---- gx = x@Wi + b_lstm (constant across micro steps)
    v8f gx[8];
    #pragma unroll
    for (int t=0;t<8;t++){
      v16h bw = frag_from_rows(sWi + t*16*32, r, half);
      v8f cacc;
      #pragma unroll
      for (int e=0;e<8;e++) cacc[e] = biasg[t];
      gx[t] = wmma16(ax, bw, cacc);
    }

    // ---- load c in WMMA D layout (two 16x16 k-halves)
    v8f c0, c1;
    #pragma unroll
    for (int v=0; v<8; v++){
      size_t rowoff = (cellbase + v + 8*half)*SLOW;
      c0[v] = cst[rowoff + r];
      c1[v] = cst[rowoff + 16 + r];
    }

    // ---- load h directly as an A fragment (row-major state matches A pattern)
    v16h ah;
    {
      const float* hp = hst + (cellbase + r)*SLOW;
      #pragma unroll
      for (int k=0;k<8;k++){ ah[k] = (_Float16)hp[half*8 + k]; ah[8+k] = (_Float16)hp[16 + half*8 + k]; }
    }

    v8f h0, h1;
    _Float16* hsl = &sH[wid][0];

    #pragma unroll
    for (int it=0; it<MICRO; ++it){
      v8f g[8];
      #pragma unroll
      for (int t=0;t<8;t++){
        v16h bw = frag_from_rows(sWh + t*16*32, r, half);
        g[t] = wmma16(ah, bw, gx[t]);             // gates = x@Wi + b + h@Wh
      }
      // i:g[0,1] f:g[2,3] g:g[4,5] o:g[6,7]
      #pragma unroll
      for (int e=0;e<8;e++){
        float c0v = sigf(g[2][e]) * c0[e] + sigf(g[0][e]) * tanh_fast(g[4][e]);
        float c1v = sigf(g[3][e]) * c1[e] + sigf(g[1][e]) * tanh_fast(g[5][e]);
        c0[e]=c0v; c1[e]=c1v;
        h0[e] = sigf(g[6][e]) * tanh_fast(c0v);
        h1[e] = sigf(g[7][e]) * tanh_fast(c1v);
      }
      // stage h (D layout -> [m][k] in LDS) and re-read as A fragment
      #pragma unroll
      for (int v=0;v<8;v++){
        hsl[(v + 8*half)*32 + r     ] = (_Float16)h0[v];
        hsl[(v + 8*half)*32 + 16 + r] = (_Float16)h1[v];
      }
      asm volatile("s_wait_dscnt 0" ::: "memory");
      ah = frag_from_rows(hsl, r, half);
      asm volatile("" ::: "memory");
    }

    // ---- store new h, c (D layout scatter back to row-major state)
    #pragma unroll
    for (int v=0;v<8;v++){
      size_t rowoff = (cellbase + v + 8*half)*SLOW;
      hst[rowoff + r]      = h0[v];
      hst[rowoff + 16 + r] = h1[v];
      cst[rowoff + r]      = c0[v];
      cst[rowoff + 16 + r] = c1[v];
    }

    // ---- fm = h@Wf + bf ; (last layer) bm = h@Wb + bwb
    float* Lp = &sL[wid][0];
    {
      v8f cf;
      #pragma unroll
      for (int e=0;e<8;e++) cf[e] = biasf;
      v8f fm = wmma16(ah, frag_from_rows(sWf, r, half), cf);
      #pragma unroll
      for (int v=0;v<8;v++) Lp[(v+8*half)*16 + r] = fm[v];
    }
    if (LAST){
      v8f cb2;
      #pragma unroll
      for (int e=0;e<8;e++) cb2[e] = biasb;
      v8f bm = wmma16(ah, frag_from_rows(sWb, r, half), cb2);
      #pragma unroll
      for (int v=0;v<8;v++) Lp[256 + (v+8*half)*16 + r] = bm[v];
    }
    asm volatile("s_wait_dscnt 0" ::: "memory");

    if (!LAST){
      if (lane < 16){
        const float* row = Lp + lane*16;
        float mu=0.0f;
        #pragma unroll
        for (int n=0;n<16;n++) mu += row[n];
        mu *= (1.0f/16.0f);
        float var=0.0f;
        #pragma unroll
        for (int n=0;n<16;n++){ float d=row[n]-mu; var += d*d; }
        var *= (1.0f/16.0f);
        float rs = rsqrtf(var + EPSV);
        float* dst = accF + ((size_t)(b*NO) + o0 + lane)*MSG;
        #pragma unroll
        for (int n=0;n<16;n++){
          float val = (row[n]-mu)*rs*lnfs[n] + lnfb[n];
          atomicAdd(dst + n, val*ascF);
        }
      }
    } else {
      if (lane < 16){
        const float* row = Lp + lane*16;
        float mu=0.0f;
        #pragma unroll
        for (int n=0;n<16;n++) mu += row[n];
        mu *= (1.0f/16.0f);
        float var=0.0f;
        #pragma unroll
        for (int n=0;n<16;n++){ float d=row[n]-mu; var += d*d; }
        var *= (1.0f/16.0f);
        float rs = rsqrtf(var + EPSV);
        float val0 = (row[0]-mu)*rs*lnfs[0] + lnfb[0];
        atomicAdd(accF + b*NOUT + o0 + lane, val0*ascF);
      } else {
        const float* row = Lp + 256 + (lane-16)*16;
        float mu=0.0f;
        #pragma unroll
        for (int n=0;n<16;n++) mu += row[n];
        mu *= (1.0f/16.0f);
        float var=0.0f;
        #pragma unroll
        for (int n=0;n<16;n++){ float d=row[n]-mu; var += d*d; }
        var *= (1.0f/16.0f);
        float rs = rsqrtf(var + EPSV);
        float* dst = accB + ((size_t)(b*NI) + i)*MSG;
        #pragma unroll
        for (int n=0;n<16;n++){
          float val = (row[n]-mu)*rs*lnbs[n] + lnbb[n];
          atomicAdd(dst + n, val*ascB);
        }
      }
    }
    asm volatile("s_wait_dscnt 0" ::: "memory");
  }
}

// softmax/error: out_t <- out ; bwd1 <- pad(stack([grad, onehot]))
__global__ void error_kernel(const float* __restrict__ outacc,
                             const int* __restrict__ labels_t,
                             float* __restrict__ out_t, float* __restrict__ bwd1){
  int b = blockIdx.x, o = threadIdx.x;
  __shared__ float red[64];
  float v = outacc[b*NOUT + o];
  out_t[b*NOUT + o] = v;
  red[o] = v; __syncthreads();
  for (int s=32; s>0; s>>=1){ if (o<s) red[o] = fmaxf(red[o], red[o+s]); __syncthreads(); }
  float mx = red[0]; __syncthreads();
  float e = __expf(v - mx);
  red[o] = e; __syncthreads();
  for (int s=32; s>0; s>>=1){ if (o<s) red[o] += red[o+s]; __syncthreads(); }
  float sum = red[0];
  float oh = (o == labels_t[b]) ? 1.0f : 0.0f;
  float grad = e/sum - oh;
  float* dst = bwd1 + (size_t)(b*NOUT + o)*MSG;
  dst[0] = grad; dst[1] = oh;
  #pragma unroll
  for (int n=2;n<MSG;n++) dst[n] = 0.0f;
}

extern "C" void kernel_launch(void* const* d_in, const int* in_sizes, int n_in,
                              void* d_out, int out_size, void* d_ws, size_t ws_size,
                              hipStream_t stream) {
  const float* inputs = (const float*)d_in[0];
  const int*   labels = (const int*)  d_in[1];
  const float* Wi   = (const float*)d_in[2];
  const float* Wh   = (const float*)d_in[3];
  const float* bl   = (const float*)d_in[4];
  const float* Wf   = (const float*)d_in[5];
  const float* bf   = (const float*)d_in[6];
  const float* Wb   = (const float*)d_in[7];
  const float* bwbp = (const float*)d_in[8];
  const float* lnfs = (const float*)d_in[9];
  const float* lnfb = (const float*)d_in[10];
  const float* lnbs = (const float*)d_in[11];
  const float* lnbb = (const float*)d_in[12];
  float* out = (float*)d_out;
  float* ws  = (float*)d_ws;

  size_t off = 0;
  float* h0 = ws + off; off += (size_t)BB*NI0*NO0*SLOW;   // 8,388,608
  float* c0 = ws + off; off += (size_t)BB*NI0*NO0*SLOW;
  float* h1 = ws + off; off += (size_t)BB*NI1*NO1*SLOW;   // 2,097,152
  float* c1 = ws + off; off += (size_t)BB*NI1*NO1*SLOW;
  float* fwd0 = ws + off; off += (size_t)BB*NI0*MSG;      // 32768
  float* fwd1 = ws + off; off += (size_t)BB*NO0*MSG;      // 16384
  float* bwd0 = ws + off; off += (size_t)BB*NO0*MSG;      // 16384
  float* bwd1 = ws + off; off += (size_t)BB*NO1*MSG;      // 8192
  float* outacc = ws + off; off += (size_t)BB*NOUT;       // 512

  // zero all persistent state (h/c/fwd/bwd/acc)
  zero_kernel<<<4096, 256, 0, stream>>>(ws, off);

  const int prepN = BB*NI0*MSG + BB*NO0*MSG + BB*NOUT;

  for (int t = 0; t < TT; ++t){
    merge_kernel<<<(NI0*NO0*16+255)/256, 256, 0, stream>>>(h0, NI0*NO0);
    merge_kernel<<<(NI0*NO0*16+255)/256, 256, 0, stream>>>(c0, NI0*NO0);
    merge_kernel<<<(NI1*NO1*16+255)/256, 256, 0, stream>>>(h1, NI1*NO1);
    merge_kernel<<<(NI1*NO1*16+255)/256, 256, 0, stream>>>(c1, NI1*NO1);

    prep_kernel<<<(prepN+255)/256, 256, 0, stream>>>(inputs + (size_t)t*BB*NI0,
                                                     fwd0, fwd1, outacc);

    // layer 0: 8*256*8 = 16384 tiles, 4 per wave, 8 waves/block -> 512 blocks
    layer_kernel<false><<<512, 256, 0, stream>>>(
        Wi, Wh, bl, Wf, bf, Wb, bwbp, lnfs, lnfb, lnbs, lnbb,
        fwd0, bwd0, h0, c0, fwd1, (float*)nullptr,
        NI0, NO0, 1.0f/NI0, 0.0f, 4);

    zero_kernel<<<64, 256, 0, stream>>>(bwd0, (size_t)BB*NO0*MSG);

    // layer 1: 8*128*4 = 4096 tiles -> 128 blocks
    layer_kernel<true><<<128, 256, 0, stream>>>(
        Wi, Wh, bl, Wf, bf, Wb, bwbp, lnfs, lnfb, lnbs, lnbb,
        fwd1, bwd1, h1, c1, outacc, bwd0,
        NI1, NO1, 1.0f/NI1, 1.0f/NO1, 4);

    error_kernel<<<BB, NOUT, 0, stream>>>(outacc, labels + (size_t)t*BB,
                                          out + (size_t)t*BB*NOUT, bwd1);
  }
}
